// ChebyKAN_7370163880754
// MI455X (gfx1250) — matrix operand order repaired and checked by reading, not compiled, not run on hardware
//
#include <hip/hip_runtime.h>
#include <hip/hip_bf16.h>
#include <cstdint>

// ---------------------------------------------------------------------------
// ChebyKAN forward as BF16 WMMA GEMM on gfx1250 (MI455X, wave32).
//   out[b,j] = sum_{i,d} T_d(tanh(x[b,i])) * C[i,j,d]
// Flatten (d,i) -> K = 9*768 = 6912:  out = A[8192 x 6912] @ Bt[6912 x 768]
//   A[b][d*768+i]  = T_d(tanh(x[b,i]))        (bf16, workspace, 113 MB, L2-resident)
//   Bt[j][d*768+i] = C[i][j][d]               (bf16, workspace, 10.6 MB)
// GEMM: block tile 128x128, 8 waves, each wave 2x4 WMMA tiles (32x64),
//       v_wmma_f32_16x16x32_bf16, double-buffered LDS, K-steps of 32.
// ---------------------------------------------------------------------------

#define M_DIM 8192
#define I_DIM 768
#define N_DIM 768
#define NDEG 9                 // D+1
#define K_DIM (I_DIM * NDEG)   // 6912
#define BM 128
#define BN 128
#define KC 32
#define LDA 40                 // halves; 80B pitch -> 16B aligned, bank-conflict-free
#define LDB 40

typedef __attribute__((ext_vector_type(16))) __bf16 v16bf;
typedef __attribute__((ext_vector_type(8)))  float  v8f;
typedef __attribute__((ext_vector_type(8)))  unsigned int v8u;

__device__ __forceinline__ uint16_t f2bf(float f) {
    // round-to-nearest-even float -> bf16
    uint32_t u = __float_as_uint(f);
    uint32_t r = u + 0x7FFFu + ((u >> 16) & 1u);
    return (uint16_t)(r >> 16);
}

// ---- Pass 1: basis A[b][d*768+i] = T_d(tanh(x[b,i])) in bf16 --------------
__global__ __launch_bounds__(256) void cheby_kernel(const float* __restrict__ x,
                                                    uint16_t* __restrict__ A) {
    int idx = blockIdx.x * 256 + threadIdx.x;      // over 8192*768, exact
    int b = idx / I_DIM;
    int i = idx - b * I_DIM;
    float t = tanhf(x[idx]);
    uint16_t* base = A + (size_t)b * K_DIM + i;    // d-major: +d*768, coalesced in i
    base[0]        = f2bf(1.0f);
    base[I_DIM]    = f2bf(t);
    float tkm1 = 1.0f, tk = t;
#pragma unroll
    for (int d = 2; d < NDEG; ++d) {
        float tn = 2.0f * t * tk - tkm1;
        tkm1 = tk; tk = tn;
        base[d * I_DIM] = f2bf(tn);
    }
}

// ---- Pass 1b: Bt[j][d*768+i] = C[i][j][d] in bf16 -------------------------
__global__ __launch_bounds__(256) void coef_kernel(const float* __restrict__ C,
                                                   uint16_t* __restrict__ Bt) {
    int idx = blockIdx.x * 256 + threadIdx.x;      // over 768*6912, exact
    int j = idx / K_DIM;
    int k = idx - j * K_DIM;
    int d = k / I_DIM;
    int i = k - d * I_DIM;
    Bt[idx] = f2bf(C[(size_t)i * (N_DIM * NDEG) + (size_t)j * NDEG + d]);
}

// ---- LDS fill: one 128x32 A chunk + 128x32 B chunk, 256 threads -----------
__device__ __forceinline__ void fill_tiles(const uint16_t* __restrict__ A,
                                           const uint16_t* __restrict__ Bt,
                                           uint16_t (*Al)[LDA], uint16_t (*Bl)[LDB],
                                           int m0, int n0, int k0, int tid) {
    int r = tid >> 2;                 // 0..63
    int p = (tid & 3) * 8;            // half-offset 0,8,16,24 (16B segments)
    *(uint4*)(&Al[r][p])      = *(const uint4*)(A  + (size_t)(m0 + r)      * K_DIM + k0 + p);
    *(uint4*)(&Al[r + 64][p]) = *(const uint4*)(A  + (size_t)(m0 + r + 64) * K_DIM + k0 + p);
    *(uint4*)(&Bl[r][p])      = *(const uint4*)(Bt + (size_t)(n0 + r)      * K_DIM + k0 + p);
    *(uint4*)(&Bl[r + 64][p]) = *(const uint4*)(Bt + (size_t)(n0 + r + 64) * K_DIM + k0 + p);
}

// ---- Pass 2: BF16 WMMA GEMM ----------------------------------------------
__global__ __launch_bounds__(256) void gemm_kernel(const uint16_t* __restrict__ A,
                                                   const uint16_t* __restrict__ Bt,
                                                   float* __restrict__ out) {
    __shared__ __align__(16) uint16_t Al[2][BM][LDA];
    __shared__ __align__(16) uint16_t Bl[2][BN][LDB];

    const int tid  = threadIdx.x;
    const int lane = tid & 31;
    const int wave = tid >> 5;        // 0..7
    const int wm   = wave >> 1;       // 0..3  -> 32-row stripe
    const int wn   = wave & 1;        // 0..1  -> 64-col stripe
    const int m0   = blockIdx.x * BM;
    const int n0   = blockIdx.y * BN;
    const int row  = lane & 15;
    const bool lo16 = (lane < 16);

    v8f acc[2][4] = {};

    fill_tiles(A, Bt, Al[0], Bl[0], m0, n0, 0, tid);
    __syncthreads();

    const int NKC = K_DIM / KC;       // 216
    for (int kc = 0; kc < NKC; ++kc) {
        const int cur = kc & 1;
        if (kc + 1 < NKC)
            fill_tiles(A, Bt, Al[1 - cur], Bl[1 - cur], m0, n0, (kc + 1) * KC, tid);

        // A fragments: ISA 16-bit A layout (lane<16: K0-7,K16-23; lane>=16: K8-15,K24-31)
        v16bf afr[2];
#pragma unroll
        for (int mi = 0; mi < 2; ++mi) {
            const uint16_t* rp = &Al[cur][wm * 32 + mi * 16 + row][0];
            const int kof = lo16 ? 0 : 8;
            uint4 l = *(const uint4*)(rp + kof);
            uint4 h = *(const uint4*)(rp + kof + 16);
            union { v8u u; v16bf b; } u;
            u.u = (v8u){l.x, l.y, l.z, l.w, h.x, h.y, h.z, h.w};
            afr[mi] = u.b;
        }
        // B fragments: same K order as A (lane<16: K0-7,K16-23; lane>=16: K8-15,K24-31), col-major LDS
        v16bf bfr[4];
#pragma unroll
        for (int ni = 0; ni < 4; ++ni) {
            const uint16_t* cp = &Bl[cur][wn * 64 + ni * 16 + row][lo16 ? 0 : 8];
            uint4 l = *(const uint4*)(cp);
            uint4 h = *(const uint4*)(cp + 16);
            union { v8u u; v16bf b; } u;
            u.u = (v8u){l.x, l.y, l.z, l.w, h.x, h.y, h.z, h.w};
            bfr[ni] = u.b;
        }

#pragma unroll
        for (int mi = 0; mi < 2; ++mi)
#pragma unroll
            for (int ni = 0; ni < 4; ++ni)
                acc[mi][ni] = __builtin_amdgcn_wmma_f32_16x16x32_bf16(
                    false, afr[mi], false, bfr[ni],
                    (short)0, acc[mi][ni], false, false);

        __syncthreads();
    }

    // C/D layout: VGPR r -> (lane<16: M=r, N=lane) / (lane>=16: M=8+r, N=lane-16)
#pragma unroll
    for (int mi = 0; mi < 2; ++mi) {
#pragma unroll
        for (int ni = 0; ni < 4; ++ni) {
            const int col   = n0 + wn * 64 + ni * 16 + row;
            const int rbase = m0 + wm * 32 + mi * 16 + (lo16 ? 0 : 8);
#pragma unroll
            for (int r = 0; r < 8; ++r)
                out[(size_t)(rbase + r) * N_DIM + col] = acc[mi][ni][r];
        }
    }
}

extern "C" void kernel_launch(void* const* d_in, const int* in_sizes, int n_in,
                              void* d_out, int out_size, void* d_ws, size_t ws_size,
                              hipStream_t stream) {
    (void)in_sizes; (void)n_in; (void)out_size; (void)ws_size;
    const float* x    = (const float*)d_in[0];   // [8192, 768] f32
    const float* coef = (const float*)d_in[1];   // [768, 768, 9] f32
    float* out        = (float*)d_out;           // [8192, 768] f32

    uint16_t* Aws = (uint16_t*)d_ws;                       // 8192*6912 bf16 = 113 MB
    uint16_t* Bws = Aws + (size_t)M_DIM * K_DIM;           // 768*6912  bf16 = 10.6 MB

    cheby_kernel<<<(M_DIM * I_DIM) / 256, 256, 0, stream>>>(x, Aws);
    coef_kernel <<<(N_DIM * K_DIM) / 256, 256, 0, stream>>>(coef, Bws);
    gemm_kernel <<<dim3(M_DIM / BM, N_DIM / BN), 256, 0, stream>>>(Aws, Bws, out);
}
